// CausalSelfAttentionWithCache_240518168728
// MI455X (gfx1250) — matrix operand order repaired and checked by reading, not compiled, not run on hardware
//
#include <hip/hip_runtime.h>

typedef unsigned short u16;
typedef __attribute__((ext_vector_type(16))) __bf16 v16bf;
typedef __attribute__((ext_vector_type(8)))  __bf16 v8bf;
typedef __attribute__((ext_vector_type(8)))  float  v8f;
typedef __attribute__((address_space(3)))    u16    lds_u16;

union Frag16 { v16bf v; v8bf h[2]; };

#define NHEAD      16
#define HDIM       128
#define NEMBD      2048
#define PREFIX_T   11
#define TOTAL_T    1035
#define SINK_T     1035
#define MAX_ATTN_T 4140
#define CACHE_T    5175
#define KPAD       4160      // 4140 rounded up to multiple of 32
#define BATCH      2
#define QK_SCALE   0.088388347648318447f  // 1/sqrt(128)

// ---------------- helpers ----------------
__device__ inline u16 f2bf(float f) {             // RNE float->bf16
    union { float f; unsigned u; } v; v.f = f;
    unsigned r = v.u + 0x7FFFu + ((v.u >> 16) & 1u);
    return (u16)(r >> 16);
}
__device__ inline unsigned pack2(float a, float b) {
    return (unsigned)f2bf(a) | ((unsigned)f2bf(b) << 16);
}
__device__ inline unsigned lds_off(const u16* p) {   // generic -> LDS byte offset
    return (unsigned)(uintptr_t)(lds_u16*)p;
}
__device__ inline void async_copy_b128(unsigned ldsoff, const u16* g) {
    asm volatile("global_load_async_to_lds_b128 %0, %1, off"
                 :: "v"(ldsoff), "v"(g) : "memory");
}
__device__ inline void wait_async0(void) {
    asm volatile("s_wait_asynccnt 0" ::: "memory");
}
__device__ inline v8f wmma_bf16(const Frag16& a, const Frag16& b, v8f c) {
    return __builtin_amdgcn_wmma_f32_16x16x32_bf16(
        false, a.v, false, b.v, (short)0, c, false, false);
}
// Reference cache-roll logic, evaluated on device from the int scalars.
__device__ inline void get_map(int cs_, int ge_, int le_,
                               int& klen, int& nold, int& cstart, int& nev) {
    int cur_end = cs_ + TOTAL_T;
    int nle;
    if (cur_end > ge_ && TOTAL_T + le_ > CACHE_T) {
        nev = TOTAL_T + le_ - CACHE_T;
        nle = le_ + (cur_end - ge_) - nev;
    } else {
        nev = 0;
        nle = le_ + (cur_end - ge_);
    }
    cstart = nle - MAX_ATTN_T; if (cstart < 0) cstart = 0;
    klen = nle - cstart;          // number of effective keys (<= 4140)
    nold = klen - TOTAL_T;        // old-cache keys; new tokens occupy [nold, klen)
}

// ---------------- fp32 -> bf16 bulk convert ----------------
__global__ __launch_bounds__(256) void cvt_kernel(const float* __restrict__ src,
                                                  u16* __restrict__ dst, int n2) {
    int i = blockIdx.x * 256 + threadIdx.x;
    int stride = gridDim.x * 256;
    for (; i < n2; i += stride) {
        float2 v = *(const float2*)(src + 2 * (size_t)i);
        *(unsigned*)(dst + 2 * (size_t)i) = pack2(v.x, v.y);
    }
}

// ---------------- bf16 GEMM: out[m,n] = sum_k A[m,k] * W[n,k] ----------------
// Wave tile 32x64, WG tile 128x128.  K-loop unrolled by 2 with two A-frag
// register sets (no pipeline copies); W tiles staged double-buffered in LDS
// via async DMA; all 4 B-frags loaded before the 8-WMMA burst.  K % 64 == 0.
__device__ inline void gemm_load_a(const u16* Ar0, const u16* Ar1, int kb, int hi,
                                   Frag16& a0, Frag16& a1) {
    a0.h[0] = *(const v8bf*)(Ar0 + kb + hi * 8);
    a0.h[1] = *(const v8bf*)(Ar0 + kb + 16 + hi * 8);
    a1.h[0] = *(const v8bf*)(Ar1 + kb + hi * 8);
    a1.h[1] = *(const v8bf*)(Ar1 + kb + 16 + hi * 8);
}
__device__ inline void gemm_mma_step(const u16* lbase, int wx, int lo, int hi,
                                     const Frag16& a0, const Frag16& a1,
                                     v8f acc[2][4]) {
    Frag16 bf[4];
#pragma unroll
    for (int t = 0; t < 4; ++t) {
        const u16* bp = lbase + (wx * 64 + t * 16 + lo) * 32 + hi * 8;
        bf[t].h[0] = *(const v8bf*)bp;
        bf[t].h[1] = *(const v8bf*)(bp + 16);
    }
#pragma unroll
    for (int t = 0; t < 4; ++t) {
        acc[0][t] = wmma_bf16(a0, bf[t], acc[0][t]);
        acc[1][t] = wmma_bf16(a1, bf[t], acc[1][t]);
    }
}
__global__ __launch_bounds__(256) void gemm_bf16_kernel(
        const u16* __restrict__ A, const u16* __restrict__ W,
        float* __restrict__ out, int M, int N, int K) {
    __shared__ u16 lB[2][128 * 32];
    int wave = threadIdx.x >> 5, lane = threadIdx.x & 31;
    int lo = lane & 15, hi = lane >> 4;
    int wy = wave >> 1, wx = wave & 1;          // 4 x 2 wave grid
    int mbase = blockIdx.y * 128 + wy * 32;
    int nwg = blockIdx.x * 128;
    int m0 = mbase + lo;      if (m0 > M - 1) m0 = M - 1;
    int m1 = mbase + 16 + lo; if (m1 > M - 1) m1 = M - 1;
    const u16* Ar0 = A + (size_t)m0 * K;
    const u16* Ar1 = A + (size_t)m1 * K;
    // staging map: thread -> (row sr, 16-col chunk sc), 32B per thread per step
    int sr = threadIdx.x >> 1;
    int sc = (threadIdx.x & 1) * 16;
    const u16* Wst = W + (size_t)(nwg + sr) * K + sc;
    unsigned lb0 = lds_off(&lB[0][sr * 32 + sc]);
    unsigned lb1 = lds_off(&lB[1][sr * 32 + sc]);

    async_copy_b128(lb0,      Wst);             // stage k-block 0
    async_copy_b128(lb0 + 16, Wst + 8);
    Frag16 aA0, aA1, aB0, aB1;
    gemm_load_a(Ar0, Ar1, 0, hi, aA0, aA1);
    wait_async0();
    __syncthreads();

    v8f acc[2][4] = {};
    for (int kb = 0; kb < K; kb += 64) {
        // half 1: compute lB[0]/aA while staging lB[1] (kb+32) + loading aB
        async_copy_b128(lb1,      Wst + kb + 32);
        async_copy_b128(lb1 + 16, Wst + kb + 40);
        gemm_load_a(Ar0, Ar1, kb + 32, hi, aB0, aB1);
        gemm_mma_step(&lB[0][0], wx, lo, hi, aA0, aA1, acc);
        wait_async0();
        __syncthreads();
        // half 2: compute lB[1]/aB while staging lB[0] (kb+64) + loading aA
        bool more = (kb + 64) < K;
        if (more) {
            async_copy_b128(lb0,      Wst + kb + 64);
            async_copy_b128(lb0 + 16, Wst + kb + 72);
            gemm_load_a(Ar0, Ar1, kb + 64, hi, aA0, aA1);
        }
        gemm_mma_step(&lB[1][0], wx, lo, hi, aB0, aB1, acc);
        if (more) {
            wait_async0();
            __syncthreads();
        }
    }
#pragma unroll
    for (int g = 0; g < 2; ++g)
#pragma unroll
        for (int t = 0; t < 4; ++t) {
            int col = nwg + wx * 64 + t * 16 + lo;
#pragma unroll
            for (int r = 0; r < 8; ++r) {
                int rowi = mbase + g * 16 + r + 8 * hi;  // C/D: M = r + 8*(lane/16)
                if (rowi < M) out[(size_t)rowi * N + col] = acc[g][t][r];
            }
        }
}

// ---------------- LayerNorm + RoPE + pack Q / new-K ----------------
__global__ __launch_bounds__(256) void ln_rope_kernel(
        const float* __restrict__ qraw, const float* __restrict__ kraw,
        const float* __restrict__ qw, const float* __restrict__ qb,
        const float* __restrict__ kw, const float* __restrict__ kbias,
        u16* __restrict__ qbuf, u16* __restrict__ keff,
        const int* pcs, const int* pge, const int* ple) {
    __shared__ float redA[8], redB[8];
    int klen, nold, cstart, nev;
    get_map(*pcs, *pge, *ple, klen, nold, cstart, nev);
    int row = blockIdx.x;
    int b = row / TOTAL_T, t = row % TOTAL_T;
    int tid = threadIdx.x, wave = tid >> 5, lane = tid & 31;
    int jnew = nold + t;
    bool img = (t >= PREFIX_T);
    float tx = 0.f, ty = 0.f;
    if (img) { int ti = t - PREFIX_T; tx = (float)(ti & 31); ty = (float)(ti >> 5); }

    for (int which = 0; which < 2; ++which) {
        const float* src = (which ? kraw : qraw) + (size_t)row * NEMBD;
        const float* wgt = which ? kw : qw;
        const float* bia = which ? kbias : qb;
        float2 vals[4];
        float s = 0.f, ss = 0.f;
#pragma unroll
        for (int i = 0; i < 4; ++i) {
            int p = tid + 256 * i;
            float2 v = *(const float2*)(src + 2 * p);
            vals[i] = v; s += v.x + v.y; ss += v.x * v.x + v.y * v.y;
        }
#pragma unroll
        for (int off = 16; off > 0; off >>= 1) {
            s  += __shfl_xor(s,  off);
            ss += __shfl_xor(ss, off);
        }
        if (lane == 0) { redA[wave] = s; redB[wave] = ss; }
        __syncthreads();
        float S = 0.f, SS = 0.f;
#pragma unroll
        for (int w = 0; w < 8; ++w) { S += redA[w]; SS += redB[w]; }
        __syncthreads();   // red[] reused by next phase
        float mu = S * (1.0f / NEMBD);
        float rstd = rsqrtf(SS * (1.0f / NEMBD) - mu * mu + 1e-5f);
#pragma unroll
        for (int i = 0; i < 4; ++i) {
            int p = tid + 256 * i;          // pair index over C (1024 pairs)
            int h = p >> 6, pi = p & 63, d = 2 * (p & 63);
            float y0 = (vals[i].x - mu) * rstd * wgt[2 * p]     + bia[2 * p];
            float y1 = (vals[i].y - mu) * rstd * wgt[2 * p + 1] + bia[2 * p + 1];
            if (img) {
                int j = pi & 31;
                float coord = (pi < 32) ? tx : ty;
                float freq = __powf(1000.0f, -(float)j * (1.0f / 32.0f));
                float sn, cn; __sincosf(coord * freq, &sn, &cn);
                float o0 = y0 * cn - y1 * sn;
                float o1 = y0 * sn + y1 * cn;
                y0 = o0; y1 = o1;
            }
            if (which == 0) {
                y0 *= QK_SCALE; y1 *= QK_SCALE;   // fold 1/sqrt(D) into Q
                unsigned* dst = (unsigned*)(qbuf +
                    (((size_t)(b * NHEAD + h)) * TOTAL_T + t) * HDIM + d);
                *dst = pack2(y0, y1);
            } else {
                unsigned* dst = (unsigned*)(keff +
                    (((size_t)(b * NHEAD + h)) * KPAD + jnew) * HDIM + d);
                *dst = pack2(y0, y1);
            }
        }
    }
}

// ---------------- assemble effective K (old part) and V-transposed ----------------
__global__ __launch_bounds__(256) void assemble_kernel(
        const float* __restrict__ cache_k, const float* __restrict__ cache_v,
        const float* __restrict__ vraw,
        u16* __restrict__ keff, u16* __restrict__ vT,
        const int* pcs, const int* pge, const int* ple) {
    int klen, nold, cstart, nev;
    get_map(*pcs, *pge, *ple, klen, nold, cstart, nev);
    int j = blockIdx.x, b = blockIdx.y, tid = threadIdx.x;
    if (j < nold) {
        int p = cstart + j;
        int src = (p < SINK_T) ? p : p + nev;   // unified: nev==0 in no-evict branch
        const float* ck = cache_k + ((size_t)b * CACHE_T + src) * NEMBD;
        const float* cv = cache_v + ((size_t)b * CACHE_T + src) * NEMBD;
#pragma unroll
        for (int i = 0; i < 8; ++i) {
            int c = tid + 256 * i; int h = c >> 7, d = c & 127;
            keff[(((size_t)(b * NHEAD + h)) * KPAD + j) * HDIM + d] = f2bf(ck[c]);
            vT[(((size_t)(b * NHEAD + h)) * HDIM + d) * KPAD + j]   = f2bf(cv[c]);
        }
    } else if (j < klen) {          // new V tokens (new K written by ln_rope)
        int t = j - nold;
        const float* vr = vraw + ((size_t)b * TOTAL_T + t) * NEMBD;
#pragma unroll
        for (int i = 0; i < 8; ++i) {
            int c = tid + 256 * i; int h = c >> 7, d = c & 127;
            vT[(((size_t)(b * NHEAD + h)) * HDIM + d) * KPAD + j] = f2bf(vr[c]);
        }
    } else {                        // zero the padding tail
#pragma unroll
        for (int i = 0; i < 8; ++i) {
            int c = tid + 256 * i; int h = c >> 7, d = c & 127;
            keff[(((size_t)(b * NHEAD + h)) * KPAD + j) * HDIM + d] = 0;
            vT[(((size_t)(b * NHEAD + h)) * HDIM + d) * KPAD + j]   = 0;
        }
    }
}

// ---------------- fused flash attention ----------------
// All 8 waves of a WG share one (b,h): K/V blocks are staged cooperatively in
// double-buffered LDS via async DMA (8x fewer VMEM requests), one barrier per
// 32-key block.  Per wave: 16 queries; S^T = K.Q^T so softmax stats are
// lane-local; O^T = V^T.P^T with P re-laid out through per-wave LDS.
__global__ __launch_bounds__(256) void attn_kernel(
        const u16* __restrict__ qbuf, const u16* __restrict__ keff,
        const u16* __restrict__ vT, u16* __restrict__ attout,
        const int* pcs, const int* pge, const int* ple) {
    __shared__ u16 lK[2][32 * 128];     // [key 0..31][d 0..127]
    __shared__ u16 lV[2][128 * 32];     // [d 0..127][key 0..31]
    __shared__ u16 Plds[8 * 16 * 32];   // per-wave 16q x 32k bf16 tile
    int klen, nold, cstart, nev;
    get_map(*pcs, *pge, *ple, klen, nold, cstart, nev);
    int bh = blockIdx.x;
    int b = bh >> 4, h = bh & 15;
    int wave = threadIdx.x >> 5, lane = threadIdx.x & 31;
    int lo = lane & 15, hi = lane >> 4;
    int q0 = blockIdx.y * 128 + wave * 16;
    const u16* Q  = qbuf + (size_t)bh * TOTAL_T * HDIM;
    const u16* Kb = keff + (size_t)bh * KPAD * HDIM;
    const u16* Vb = vT   + (size_t)bh * HDIM * KPAD;
    u16* pl = &Plds[wave * 512];

    int qrow = q0 + lo; if (qrow > TOTAL_T - 1) qrow = TOTAL_T - 1;
    Frag16 qf[4];                       // Q as B-frags, one per 32-d chunk
#pragma unroll
    for (int c = 0; c < 4; ++c) {
        const u16* qp = Q + (size_t)qrow * HDIM + c * 32 + hi * 8;
        qf[c].h[0] = *(const v8bf*)qp;
        qf[c].h[1] = *(const v8bf*)(qp + 16);
    }
    // cooperative staging maps (32B per thread per tile)
    int tr = threadIdx.x >> 3, tc = (threadIdx.x & 7) * 16;  // K tile row/col
    int vr = threadIdx.x >> 1, vc = (threadIdx.x & 1) * 16;  // V tile row/col
    const u16* kg = Kb + (size_t)tr * HDIM + tc;
    const u16* vg = Vb + (size_t)vr * KPAD + vc;
    unsigned lk0 = lds_off(&lK[0][tr * 128 + tc]);
    unsigned lk1 = lds_off(&lK[1][tr * 128 + tc]);
    unsigned lv0 = lds_off(&lV[0][vr * 32 + vc]);
    unsigned lv1 = lds_off(&lV[1][vr * 32 + vc]);

    async_copy_b128(lk0,      kg);      // stage key-block 0
    async_copy_b128(lk0 + 16, kg + 8);
    async_copy_b128(lv0,      vg);
    async_copy_b128(lv0 + 16, vg + 8);

    v8f o[8] = {};                      // O^T accumulators, 8 d-tiles
    float mrow = -3.0e38f, lrow = 0.0f;
    wait_async0();
    __syncthreads();

    int cur = 0;
    for (int kb = 0; kb < klen; kb += 32) {
        bool more = (kb + 32) < klen;
        if (more) {                     // DMA next K/V block into other buffer
            unsigned lkn = cur ? lk0 : lk1;
            unsigned lvn = cur ? lv0 : lv1;
            const u16* kgn = kg + (size_t)(kb + 32) * HDIM;
            const u16* vgn = vg + (kb + 32);
            async_copy_b128(lkn,      kgn);
            async_copy_b128(lkn + 16, kgn + 8);
            async_copy_b128(lvn,      vgn);
            async_copy_b128(lvn + 16, vgn + 8);
        }
        const u16* lKc = &lK[cur][0];
        const u16* lVc = &lV[cur][0];
        v8f s[2] = {};                  // S^T for the two 16-key groups
#pragma unroll
        for (int g = 0; g < 2; ++g) {
            const u16* kr = lKc + (g * 16 + lo) * 128 + hi * 8;
#pragma unroll
            for (int c = 0; c < 4; ++c) {
                Frag16 kf;              // K rows as A-frags
                kf.h[0] = *(const v8bf*)(kr + c * 32);
                kf.h[1] = *(const v8bf*)(kr + c * 32 + 16);
                s[g] = wmma_bf16(kf, qf[c], s[g]);
            }
        }
        if (kb + 32 > klen) {           // mask tail keys
#pragma unroll
            for (int g = 0; g < 2; ++g)
#pragma unroll
                for (int r = 0; r < 8; ++r)
                    if (kb + g * 16 + hi * 8 + r >= klen) s[g][r] = -3.0e38f;
        }
        float bm = -3.0e38f;
#pragma unroll
        for (int g = 0; g < 2; ++g)
#pragma unroll
            for (int r = 0; r < 8; ++r) bm = fmaxf(bm, s[g][r]);
        bm = fmaxf(bm, __shfl_xor(bm, 16));
        float mnew = fmaxf(mrow, bm);
        float corr = __expf(mrow - mnew);
        float ls = 0.0f;
#pragma unroll
        for (int g = 0; g < 2; ++g)
#pragma unroll
            for (int r = 0; r < 8; ++r) {
                float p = __expf(s[g][r] - mnew);
                s[g][r] = p; ls += p;
            }
        ls += __shfl_xor(ls, 16);
        lrow = lrow * corr + ls;
        mrow = mnew;
        // C/D-frag -> [query][key] LDS first (DS latency overlaps rescale below)
#pragma unroll
        for (int g = 0; g < 2; ++g)
#pragma unroll
            for (int r = 0; r < 8; r += 2)
                *(unsigned*)&pl[lo * 32 + g * 16 + hi * 8 + r] =
                    pack2(s[g][r], s[g][r + 1]);
#pragma unroll
        for (int f = 0; f < 8; ++f)
#pragma unroll
            for (int r = 0; r < 8; ++r) o[f][r] *= corr;
        Frag16 pf;                      // P^T as B-frag (k = hi*16 + 0..15)
        {
            const u16* pp = &pl[lo * 32 + hi * 16];
            pf.h[0] = *(const v8bf*)pp;
            pf.h[1] = *(const v8bf*)(pp + 8);
        }
#pragma unroll
        for (int f = 0; f < 8; ++f) {
            Frag16 vf;                  // V^T rows as A-frags from LDS
            const u16* vp = lVc + (f * 16 + lo) * 32 + hi * 16;
            vf.h[0] = *(const v8bf*)vp;
            vf.h[1] = *(const v8bf*)(vp + 8);
            o[f] = wmma_bf16(vf, pf, o[f]);
        }
        if (more) wait_async0();
        __syncthreads();
        cur ^= 1;
    }
    float inv = 1.0f / lrow;
    int t = q0 + lo;
    if (t < TOTAL_T) {
        unsigned* dst = (unsigned*)(attout + ((size_t)b * TOTAL_T + t) * NEMBD + h * HDIM);
#pragma unroll
        for (int f = 0; f < 8; ++f)
#pragma unroll
            for (int r = 0; r < 8; r += 2) {
                int d = f * 16 + hi * 8 + r;
                dst[d >> 1] = pack2(o[f][r] * inv, o[f][r + 1] * inv);
            }
    }
}

// ---------------- launcher ----------------
extern "C" void kernel_launch(void* const* d_in, const int* in_sizes, int n_in,
                              void* d_out, int out_size, void* d_ws, size_t ws_size,
                              hipStream_t stream) {
    (void)in_sizes; (void)n_in; (void)out_size; (void)ws_size;
    const float* x      = (const float*)d_in[0];
    const float* Wq     = (const float*)d_in[1];
    const float* Wk     = (const float*)d_in[2];
    const float* Wv     = (const float*)d_in[3];
    const float* Wp     = (const float*)d_in[4];
    const float* qn_w   = (const float*)d_in[5];
    const float* qn_b   = (const float*)d_in[6];
    const float* kn_w   = (const float*)d_in[7];
    const float* kn_b   = (const float*)d_in[8];
    const float* cacheK = (const float*)d_in[9];
    const float* cacheV = (const float*)d_in[10];
    const int*   pcs    = (const int*)d_in[11];
    const int*   pge    = (const int*)d_in[12];
    const int*   ple    = (const int*)d_in[13];
    float* out = (float*)d_out;

    const int Mrows = BATCH * TOTAL_T;               // 2070
    const size_t szXB   = (size_t)Mrows * NEMBD * sizeof(u16);
    const size_t szW    = (size_t)NEMBD * NEMBD * sizeof(u16);
    const size_t szRaw  = (size_t)Mrows * NEMBD * sizeof(float);
    const size_t szQbuf = (size_t)BATCH * NHEAD * TOTAL_T * HDIM * sizeof(u16);
    const size_t szKeff = (size_t)BATCH * NHEAD * KPAD * HDIM * sizeof(u16);

    char* ws = (char*)d_ws;
    size_t off = 0;
    auto take = [&](size_t bytes) -> char* {
        char* p = ws + off;
        off = (off + bytes + 255) & ~(size_t)255;
        return p;
    };
    u16*   xb   = (u16*)take(szXB);
    u16*   wqb  = (u16*)take(szW);
    u16*   wkb  = (u16*)take(szW);
    u16*   wvb  = (u16*)take(szW);
    u16*   wpb  = (u16*)take(szW);
    float* qraw = (float*)take(szRaw);
    float* kraw = (float*)take(szRaw);
    float* vraw = (float*)take(szRaw);
    u16*   qbuf = (u16*)take(szQbuf);
    u16*   keff = (u16*)take(szKeff);
    u16*   vTb  = (u16*)take(szKeff);
    u16*   attb = (u16*)take(szXB);

    // 1) bf16 conversions
    cvt_kernel<<<dim3(512), dim3(256), 0, stream>>>(x,  xb,  Mrows * NEMBD / 2);
    cvt_kernel<<<dim3(512), dim3(256), 0, stream>>>(Wq, wqb, NEMBD * NEMBD / 2);
    cvt_kernel<<<dim3(512), dim3(256), 0, stream>>>(Wk, wkb, NEMBD * NEMBD / 2);
    cvt_kernel<<<dim3(512), dim3(256), 0, stream>>>(Wv, wvb, NEMBD * NEMBD / 2);
    cvt_kernel<<<dim3(512), dim3(256), 0, stream>>>(Wp, wpb, NEMBD * NEMBD / 2);

    // 2) QKV projections (WMMA bf16)
    dim3 gg(NEMBD / 128, (Mrows + 127) / 128);
    gemm_bf16_kernel<<<gg, 256, 0, stream>>>(xb, wqb, qraw, Mrows, NEMBD, NEMBD);
    gemm_bf16_kernel<<<gg, 256, 0, stream>>>(xb, wkb, kraw, Mrows, NEMBD, NEMBD);
    gemm_bf16_kernel<<<gg, 256, 0, stream>>>(xb, wvb, vraw, Mrows, NEMBD, NEMBD);

    // 3) LN + RoPE + pack Q / new K
    ln_rope_kernel<<<dim3(Mrows), 256, 0, stream>>>(qraw, kraw, qn_w, qn_b, kn_w, kn_b,
                                                    qbuf, keff, pcs, pge, ple);
    // 4) cache roll -> effective K, V^T (+ zero padding tail)
    assemble_kernel<<<dim3(KPAD, BATCH), 256, 0, stream>>>(cacheK, cacheV, vraw,
                                                           keff, vTb, pcs, pge, ple);
    // 5) fused flash attention (WMMA bf16, async-staged K/V)
    attn_kernel<<<dim3(BATCH * NHEAD, (TOTAL_T + 127) / 128), 256, 0, stream>>>(
        qbuf, keff, vTb, attb, pcs, pge, ple);
    // 6) output projection -> f32 out
    gemm_bf16_kernel<<<gg, 256, 0, stream>>>(attb, wpb, out, Mrows, NEMBD, NEMBD);
}
